// MMRecBlock_89996744720852
// MI455X (gfx1250) — matrix operand order repaired and checked by reading, not compiled, not run on hardware
//
#include <hip/hip_runtime.h>
#include <hip/hip_bf16.h>

typedef __attribute__((ext_vector_type(16))) __bf16 bf16x16;
typedef __attribute__((ext_vector_type(8)))  float  v8f;

// ---------------------------------------------------------------- helpers
__device__ __forceinline__ float sigmf(float x){ return 1.0f/(1.0f + __expf(-x)); }
__device__ __forceinline__ float geluf(float x){
  float x3 = x*x*x;
  return 0.5f*x*(1.0f + tanhf(0.7978845608028654f*(x + 0.044715f*x3)));
}

// CDNA5 async global->LDS copy (16B per lane), tracked by ASYNCcnt.
// Generic LDS pointer's low 32 bits are the in-allocation LDS byte address.
__device__ __forceinline__ void async_g2l_b128(const void* gsrc, void* lds_dst){
  unsigned l = (unsigned)(unsigned long long)(uintptr_t)lds_dst;
  unsigned long long g = (unsigned long long)(uintptr_t)gsrc;
  asm volatile("global_load_async_to_lds_b128 %0, %1, off" :: "v"(l), "v"(g) : "memory");
}
__device__ __forceinline__ void wait_async0(){
  asm volatile("s_wait_asynccnt 0x0" ::: "memory");
}

// A-fragment (16x32 bf16): lane L holds M=L%16; K = (L/16)*16 + e
__device__ __forceinline__ bf16x16 frag_a(const __bf16* p, int stride, int row, int klo){
  union { bf16x16 v; uint4 u[2]; } t;
  const __bf16* r = p + (long)row*stride + klo;
  t.u[0] = *(const uint4*)(r);
  t.u[1] = *(const uint4*)(r + 8);
  return t.v;
}
// B-fragment (32x16 bf16), stored [n][k]: lane L holds N=L%16; K = (L/16)*16 + e
__device__ __forceinline__ bf16x16 frag_b(const __bf16* p, int stride, int row, int kb){
  union { bf16x16 v; uint4 u[2]; } t;
  const __bf16* r = p + (long)row*stride + kb;
  t.u[0] = *(const uint4*)(r);
  t.u[1] = *(const uint4*)(r + 8);
  return t.v;
}

// ---------------------------------------------------------------- GEMM
// C[M,N] = act(alpha * A[M,K] x B + bias + resid). A bf16 row-major.
// BT=0: B bf16 row-major [K,N].  BT=1: B bf16 row-major [N,K] (i.e. B^T).
// Batched via blockIdx.z: off = (z/zdiv)*s1 + (z%zdiv)*s2 per operand.
// Double-buffered LDS stages filled with async global->LDS (ASYNCcnt) copies.
template<int BT>
__global__ __launch_bounds__(256) void k_gemm(
    const __bf16* __restrict__ A, const __bf16* __restrict__ Bm,
    float* __restrict__ Cf, __bf16* __restrict__ Cb,
    const float* __restrict__ bias, const float* __restrict__ Rr,
    int M, int N, int K, int lda, int ldb, int ldc, int ldr,
    int zdiv, long sA1, long sA2, long sB1, long sB2,
    long sC1, long sC2, long sR1, long sR2,
    float alpha, int act)
{
  constexpr int LST  = 40;            // padded LDS row stride (bf16): 80B, 16B-aligned
  constexpr int ASTG = 128*LST;       // elements per stage
  __shared__ __bf16 lsA[2*ASTG];      // 2 x (128 rows x 32 K) tile
  __shared__ __bf16 lsB[2*ASTG];      // 2 x (128 n-rows x 32 K), stored [n][k]
  const int tid  = threadIdx.x;
  const int lane = tid & 31;
  const int wid  = tid >> 5;
  const int wm   = wid & 1;           // 2 waves in M (64 rows each)
  const int wn   = wid >> 1;          // 4 waves in N (32 cols each)
  const int z    = blockIdx.z;
  const long offA = (long)(z/zdiv)*sA1 + (long)(z%zdiv)*sA2;
  const long offB = (long)(z/zdiv)*sB1 + (long)(z%zdiv)*sB2;
  const int m0 = blockIdx.y * 128;
  const int n0 = blockIdx.x * 128;
  const __bf16* Ab = A + offA + (long)m0*lda;
  const __bf16* Bb = Bm + offB;

  v8f zv = {};
  v8f acc[4][2];
  #pragma unroll
  for (int mi=0; mi<4; ++mi){ acc[mi][0]=zv; acc[mi][1]=zv; }

  const int ar = tid >> 1, aseg = tid & 1;      // A/B(NT) tile: row, 16-col segment

  auto loadA = [&](int ks, int buf){
    const __bf16* src = Ab + (long)ar*lda + ks + aseg*16;
    __bf16* dst = lsA + buf*ASTG + ar*LST + aseg*16;
    async_g2l_b128(src, dst);
    async_g2l_b128(src + 8, dst + 8);
  };
  auto loadB = [&](int ks, int buf){
    if (BT) {     // B stored [N,K]: straight async copy into [n][k]
      const __bf16* src = Bb + (long)(n0 + ar)*ldb + ks + aseg*16;
      __bf16* dst = lsB + buf*ASTG + ar*LST + aseg*16;
      async_g2l_b128(src, dst);
      async_g2l_b128(src + 8, dst + 8);
    } else {      // B stored [K,N]: transpose-scatter into [n][k]
      const int bk = tid >> 3, bseg = tid & 7;  // k-row 0..31, 16-col segment
      const __bf16* src = Bb + (long)(ks+bk)*ldb + n0 + bseg*16;
      union { uint4 u[2]; __bf16 h[16]; } t;
      t.u[0] = ((const uint4*)src)[0];
      t.u[1] = ((const uint4*)src)[1];
      __bf16* dst = lsB + buf*ASTG;
      #pragma unroll
      for (int j = 0; j < 16; ++j) dst[(bseg*16 + j)*LST + bk] = t.h[j];
    }
  };

  loadA(0, 0);
  loadB(0, 0);
  int buf = 0;
  for (int ks = 0; ks < K; ks += 32, buf ^= 1) {
    wait_async0();                    // stage `buf` async fills complete (this wave)
    __syncthreads();                  // all waves' fills visible
    if (ks + 32 < K) {                // overlap next stage's fill with this WMMA burst
      loadA(ks + 32, buf ^ 1);
      loadB(ks + 32, buf ^ 1);
      __builtin_prefetch(Ab + (long)ar*lda + ks + 64, 0, 1);   // global_prefetch_b8
    }
    const int abase = (lane >> 4) << 4;
    const int bbase = (lane >> 4) << 4;
    bf16x16 af[4], bfv[2];
    #pragma unroll
    for (int mi=0; mi<4; ++mi)
      af[mi] = frag_a(lsA + buf*ASTG, LST, wm*64 + mi*16 + (lane&15), abase);
    #pragma unroll
    for (int ni=0; ni<2; ++ni)
      bfv[ni] = frag_b(lsB + buf*ASTG, LST, wn*32 + ni*16 + (lane&15), bbase);
    #pragma unroll
    for (int mi=0; mi<4; ++mi)
      #pragma unroll
      for (int ni=0; ni<2; ++ni)
        acc[mi][ni] = __builtin_amdgcn_wmma_f32_16x16x32_bf16(
            false, af[mi], false, bfv[ni], (short)0, acc[mi][ni], false, false);
  }

  const long offC = (long)(z/zdiv)*sC1 + (long)(z%zdiv)*sC2;
  const long offR = (long)(z/zdiv)*sR1 + (long)(z%zdiv)*sR2;
  #pragma unroll
  for (int mi=0; mi<4; ++mi)
    #pragma unroll
    for (int ni=0; ni<2; ++ni)
      #pragma unroll
      for (int e=0; e<8; ++e) {
        const int m = m0 + wm*64 + mi*16 + ((lane>>4)<<3) + e;
        const int n = n0 + wn*32 + ni*16 + (lane&15);
        float v = alpha * acc[mi][ni][e];
        if (bias) v += bias[n];
        if (Rr)   v += Rr[offR + (long)m*ldr + n];
        if (act == 1) v = geluf(v);
        else if (act == 2) v = sigmf(v);
        const long ci = offC + (long)m*ldc + n;
        if (Cf) Cf[ci] = v;
        if (Cb) Cb[ci] = (__bf16)v;
      }
}

// ---------------------------------------------------------------- scan (serial recurrence)
// h state in LDS: fp32 master + zero-padded 16-row bf16 copy for WMMA.
// Per step: gate_logit[16x1024] = Hb[16x1024] @ WghT^T, via 16 waves x 4 N-tiles.
__global__ __launch_bounds__(512) void k_scan(
    const float* __restrict__ zg, const float* __restrict__ zf,
    const float* __restrict__ gam, const float* __restrict__ bgh,
    const __bf16* __restrict__ WghT,    // [N=1024][K=1024] = Wgh transposed, bf16
    float* __restrict__ hseq)
{
  constexpr int Dd = 1024, Sd = 4096, HBS = 1032;
  __shared__ __bf16 hb[16*HBS];
  __shared__ float  hf[4*Dd];
  const int tid = threadIdx.x, lane = tid & 31, w = tid >> 5;  // 16 waves
  for (int i = tid; i < 16*HBS; i += 512) hb[i] = (__bf16)0.0f;
  for (int i = tid; i < 4*Dd;   i += 512) hf[i] = 0.0f;
  __syncthreads();
  const int n0 = w * 64;
  const v8f zv = {};
  for (int t = 0; t < Sd; ++t) {
    v8f acc[4]; acc[0]=zv; acc[1]=zv; acc[2]=zv; acc[3]=zv;
    for (int kt = 0; kt < 32; ++kt) {
      const int abase = kt*32 + ((lane >> 4) << 4);
      const int bbase = kt*32 + ((lane >> 4) << 4);
      bf16x16 af = frag_a(hb, HBS, lane & 15, abase);
      #pragma unroll
      for (int ni = 0; ni < 4; ++ni) {
        bf16x16 bfv = frag_b(WghT, Dd, n0 + ni*16 + (lane & 15), bbase);
        acc[ni] = __builtin_amdgcn_wmma_f32_16x16x32_bf16(
            false, af, false, bfv, (short)0, acc[ni], false, false);
      }
    }
    __syncthreads();            // phase-1 reads of hb complete
    if (lane < 16) {            // C layout: m=(lane/16)*8+e -> lanes<16, e<4 cover b=0..3
      #pragma unroll
      for (int ni = 0; ni < 4; ++ni)
        #pragma unroll
        for (int e = 0; e < 4; ++e) {
          const int b = e;
          const int d = n0 + ni*16 + lane;
          const long idx = ((long)b*Sd + t)*Dd + d;
          const float gate = sigmf(acc[ni][e] + zg[idx] + bgh[d]);
          const float hp = hf[b*Dd + d];
          const float ht = (1.0f - gate)*hp + gate*zf[idx] + gam[idx]*hp;
          hf[b*Dd + d]  = ht;
          hb[b*HBS + d] = (__bf16)ht;
          hseq[idx] = ht;
        }
    }
    __syncthreads();            // hb updated before next step's WMMA
  }
}

// ---------------------------------------------------------------- small kernels
__global__ void k_f32_to_bf16(const float* __restrict__ in, __bf16* __restrict__ out, long n){
  long i = (long)blockIdx.x*blockDim.x + threadIdx.x;
  const long st = (long)gridDim.x*blockDim.x;
  for (; i < n; i += st) out[i] = (__bf16)in[i];
}
__global__ void k_transpose_bf16(const float* __restrict__ in, __bf16* __restrict__ out, int K, int N){
  long i = (long)blockIdx.x*blockDim.x + threadIdx.x;
  const long st = (long)gridDim.x*blockDim.x;
  const long tot = (long)K*N;
  for (; i < tot; i += st) {
    const int k = (int)(i / N), n = (int)(i % N);
    out[(long)n*K + k] = (__bf16)in[i];
  }
}
__global__ __launch_bounds__(256) void k_rmsnorm(const float* __restrict__ x,
    const float* __restrict__ scale, __bf16* __restrict__ out, int Dd){
  __shared__ float red[256];
  const long row = blockIdx.x;
  const float* xr = x + row*Dd;
  float s = 0.0f;
  for (int d = threadIdx.x; d < Dd; d += 256) { const float v = xr[d]; s += v*v; }
  red[threadIdx.x] = s; __syncthreads();
  for (int wdt = 128; wdt > 0; wdt >>= 1) {
    if (threadIdx.x < wdt) red[threadIdx.x] += red[threadIdx.x + wdt];
    __syncthreads();
  }
  const float rs = rsqrtf(red[0]/(float)Dd + 1e-6f);
  for (int d = threadIdx.x; d < Dd; d += 256) out[row*Dd + d] = (__bf16)(xr[d]*scale[d]*rs);
}
__global__ __launch_bounds__(256) void k_softmax(const float* __restrict__ s,
    __bf16* __restrict__ p, int Mc){
  __shared__ float red[256];
  const long row = blockIdx.x;
  const float* sr = s + row*Mc;
  float mx = -1e30f;
  for (int j = threadIdx.x; j < Mc; j += 256) mx = fmaxf(mx, sr[j]);
  red[threadIdx.x] = mx; __syncthreads();
  for (int wdt = 128; wdt > 0; wdt >>= 1) {
    if (threadIdx.x < wdt) red[threadIdx.x] = fmaxf(red[threadIdx.x], red[threadIdx.x + wdt]);
    __syncthreads();
  }
  const float m = red[0]; __syncthreads();
  float sum = 0.0f;
  for (int j = threadIdx.x; j < Mc; j += 256) sum += __expf(sr[j] - m);
  red[threadIdx.x] = sum; __syncthreads();
  for (int wdt = 128; wdt > 0; wdt >>= 1) {
    if (threadIdx.x < wdt) red[threadIdx.x] += red[threadIdx.x + wdt];
    __syncthreads();
  }
  const float inv = 1.0f / red[0];
  for (int j = threadIdx.x; j < Mc; j += 256)
    p[row*Mc + j] = (__bf16)(__expf(sr[j] - m) * inv);
}
__global__ void k_gamma(const float* __restrict__ a, const float* __restrict__ b,
                        float* __restrict__ g, long n){
  long i = (long)blockIdx.x*blockDim.x + threadIdx.x;
  const long st = (long)gridDim.x*blockDim.x;
  for (; i < n; i += st)
    g[i] = fminf(fmaxf(sigmf(a[i])*sigmf(b[i]), 1e-6f), 1.0f - 1e-6f);
}
__global__ void k_combine(const float* __restrict__ x, const float* __restrict__ h,
                          const float* __restrict__ mo, const float* __restrict__ v,
                          float* __restrict__ xr, long n){
  long i = (long)blockIdx.x*blockDim.x + threadIdx.x;
  const long st = (long)gridDim.x*blockDim.x;
  for (; i < n; i += st) xr[i] = x[i] + h[i] + mo[i] + 0.1f*v[i];
}

// ---------------------------------------------------------------- host
extern "C" void kernel_launch(void* const* d_in, const int* in_sizes, int n_in,
                              void* d_out, int out_size, void* d_ws, size_t ws_size,
                              hipStream_t stream) {
  (void)in_sizes; (void)n_in; (void)out_size; (void)ws_size;
  constexpr int Bc=4, Sc=4096, Dc=1024, Hc=8, Fc=4096, Mm=512, Dq=256;
  constexpr int NTt = Bc*Sc;                    // 16384 token rows
  constexpr long NDl = (long)NTt*Dc;            // 16,777,216
  constexpr int NTM = Bc*Mm;                    // 2048 memory rows
  constexpr long DD = (long)Dc*Dc, DDq = (long)Dc*Dq, DF = (long)Dc*Fc;

  const float* x    = (const float*)d_in[0];
  const float* memk = (const float*)d_in[1];
  const float* memv = (const float*)d_in[2];
  const float* n1s  = (const float*)d_in[3];
  const float* n2s  = (const float*)d_in[4];
  const float* Wq=(const float*)d_in[5],  *bq=(const float*)d_in[6];
  const float* Wk=(const float*)d_in[7],  *bk=(const float*)d_in[8];
  const float* Wv=(const float*)d_in[9],  *bv=(const float*)d_in[10];
  const float* Wz=(const float*)d_in[11], *bz=(const float*)d_in[12];
  const float* Wgz=(const float*)d_in[13],*bgz=(const float*)d_in[14];
  const float* Wgh=(const float*)d_in[15],*bgh=(const float*)d_in[16];
  const float* Wg1=(const float*)d_in[17],*bg1=(const float*)d_in[18];
  const float* Wg2=(const float*)d_in[19],*bg2=(const float*)d_in[20];
  const float* Wgc1=(const float*)d_in[21],*bgc1=(const float*)d_in[22];
  const float* Wgc2=(const float*)d_in[23],*bgc2=(const float*)d_in[24];
  const float* Awq=(const float*)d_in[25],*abq=(const float*)d_in[26];
  const float* Awk=(const float*)d_in[27],*abk=(const float*)d_in[28];
  const float* Awv=(const float*)d_in[29],*abv=(const float*)d_in[30];
  const float* Awo=(const float*)d_in[31],*abo=(const float*)d_in[32];
  const float* Wf1=(const float*)d_in[33],*bf1=(const float*)d_in[34];
  const float* Wf2=(const float*)d_in[35],*bf2=(const float*)d_in[36];

  char* wsb = (char*)d_ws; size_t off = 0;
  auto alloc = [&](size_t bytes)->void* {
    void* p = wsb + off; off += (bytes + 255) & ~(size_t)255; return p;
  };
  // bf16 weights
  __bf16* wqb  = (__bf16*)alloc(DD*2);  __bf16* wkb  = (__bf16*)alloc(DD*2);
  __bf16* wvb  = (__bf16*)alloc(DD*2);  __bf16* wzb  = (__bf16*)alloc(DD*2);
  __bf16* wgzb = (__bf16*)alloc(DD*2);  __bf16* wghT = (__bf16*)alloc(DD*2);
  __bf16* wg1b = (__bf16*)alloc(DDq*2); __bf16* wg2b = (__bf16*)alloc(DDq*2);
  __bf16* wgc1b= (__bf16*)alloc(DDq*2); __bf16* wgc2b= (__bf16*)alloc(DDq*2);
  __bf16* awqb = (__bf16*)alloc(DD*2);  __bf16* awkb = (__bf16*)alloc(DD*2);
  __bf16* awvb = (__bf16*)alloc(DD*2);  __bf16* awob = (__bf16*)alloc(DD*2);
  __bf16* wf1b = (__bf16*)alloc(DF*2);  __bf16* wf2b = (__bf16*)alloc(DF*2);
  __bf16* mkb  = (__bf16*)alloc((long)NTM*Dc*2);
  __bf16* mvb  = (__bf16*)alloc((long)NTM*Dc*2);
  // activations
  __bf16* xnb   = (__bf16*)alloc(NDl*2);
  __bf16* qbf   = (__bf16*)alloc(NDl*2);
  __bf16* kbf   = (__bf16*)alloc(NDl*2);
  float*  vf    = (float*) alloc(NDl*4);
  float*  zf    = (float*) alloc(NDl*4);
  __bf16* zbf   = (__bf16*)alloc(NDl*2);
  float*  zgf   = (float*) alloc(NDl*4);
  __bf16* g1b   = (__bf16*)alloc((long)NTt*Dq*2);
  __bf16* gc1b  = (__bf16*)alloc((long)NTt*Dq*2);
  float*  glog1 = (float*) alloc(NDl*4);
  float*  glogc = (float*) alloc(NDl*4);
  float*  gammaf= (float*) alloc(NDl*4);
  float*  hseqf = (float*) alloc(NDl*4);
  __bf16* qhb   = (__bf16*)alloc(NDl*2);
  __bf16* khb   = (__bf16*)alloc((long)NTM*Dc*2);
  __bf16* vhb   = (__bf16*)alloc((long)NTM*Dc*2);
  float*  scoresf=(float*) alloc((long)Bc*Hc*Sc*Mm*4);
  __bf16* probsb =(__bf16*)alloc((long)Bc*Hc*Sc*Mm*2);
  float*  ctxf  = (float*) alloc(NDl*4);
  __bf16* ctxb  = (__bf16*)alloc(NDl*2);
  float*  mof   = (float*) alloc(NDl*4);
  float*  xresf = (float*) alloc(NDl*4);
  __bf16* xn2b  = (__bf16*)alloc(NDl*2);
  __bf16* f1b   = (__bf16*)alloc((long)NTt*Fc*2);

  auto cvt = [&](const float* s, __bf16* d, long n){
    k_f32_to_bf16<<<2048, 256, 0, stream>>>(s, d, n);
  };
  cvt(Wq,wqb,DD); cvt(Wk,wkb,DD); cvt(Wv,wvb,DD); cvt(Wz,wzb,DD); cvt(Wgz,wgzb,DD);
  k_transpose_bf16<<<2048,256,0,stream>>>(Wgh, wghT, Dc, Dc);
  cvt(Wg1,wg1b,DDq); cvt(Wg2,wg2b,DDq); cvt(Wgc1,wgc1b,DDq); cvt(Wgc2,wgc2b,DDq);
  cvt(Awq,awqb,DD); cvt(Awk,awkb,DD); cvt(Awv,awvb,DD); cvt(Awo,awob,DD);
  cvt(Wf1,wf1b,DF); cvt(Wf2,wf2b,DF);
  cvt(memk,mkb,(long)NTM*Dc); cvt(memv,mvb,(long)NTM*Dc);

  auto gemmL = [&](int BT, const __bf16* A, const __bf16* Bm, float* Cf, __bf16* Cb,
                   const float* bias, const float* Rr,
                   int M, int N, int K, int lda, int ldb, int ldc, int ldr,
                   int batches, int zdiv,
                   long sA1, long sA2, long sB1, long sB2,
                   long sC1, long sC2, long sR1, long sR2,
                   float alpha, int act){
    dim3 g(N/128, M/128, batches), blk(256);
    if (BT) k_gemm<1><<<g,blk,0,stream>>>(A,Bm,Cf,Cb,bias,Rr,M,N,K,lda,ldb,ldc,ldr,
                                          zdiv,sA1,sA2,sB1,sB2,sC1,sC2,sR1,sR2,alpha,act);
    else    k_gemm<0><<<g,blk,0,stream>>>(A,Bm,Cf,Cb,bias,Rr,M,N,K,lda,ldb,ldc,ldr,
                                          zdiv,sA1,sA2,sB1,sB2,sC1,sC2,sR1,sR2,alpha,act);
  };

  // 1. x_norm (bf16)
  k_rmsnorm<<<NTt, 256, 0, stream>>>(x, n1s, xnb, Dc);
  // 2. q,k,v,z projections
  gemmL(0, xnb, wqb, nullptr, qbf, bq, nullptr, NTt,Dc,Dc, Dc,Dc,Dc,0, 1,1, 0,0,0,0,0,0,0,0, 1.f,0);
  gemmL(0, xnb, wkb, nullptr, kbf, bk, nullptr, NTt,Dc,Dc, Dc,Dc,Dc,0, 1,1, 0,0,0,0,0,0,0,0, 1.f,0);
  gemmL(0, xnb, wvb, vf, nullptr, bv, nullptr, NTt,Dc,Dc, Dc,Dc,Dc,0, 1,1, 0,0,0,0,0,0,0,0, 1.f,0);
  gemmL(0, xnb, wzb, zf, zbf,     bz, nullptr, NTt,Dc,Dc, Dc,Dc,Dc,0, 1,1, 0,0,0,0,0,0,0,0, 1.f,0);
  // 3. zg = z@Wgz + bgz (hoisted out of the scan)
  gemmL(0, zbf, wgzb, zgf, nullptr, bgz, nullptr, NTt,Dc,Dc, Dc,Dc,Dc,0, 1,1, 0,0,0,0,0,0,0,0, 1.f,0);
  // 4. gamma = clip(sigmoid(gelu(z@Wg1+bg1)@Wg2+bg2) * sigmoid(gelu(k@Wgc1+bgc1)@Wgc2+bgc2))
  gemmL(0, zbf, wg1b, nullptr, g1b,  bg1,  nullptr, NTt,Dq,Dc, Dc,Dq,Dq,0, 1,1, 0,0,0,0,0,0,0,0, 1.f,1);
  gemmL(0, g1b, wg2b, glog1, nullptr, bg2, nullptr, NTt,Dc,Dq, Dq,Dc,Dc,0, 1,1, 0,0,0,0,0,0,0,0, 1.f,0);
  gemmL(0, kbf, wgc1b, nullptr, gc1b, bgc1, nullptr, NTt,Dq,Dc, Dc,Dq,Dq,0, 1,1, 0,0,0,0,0,0,0,0, 1.f,1);
  gemmL(0, gc1b, wgc2b, glogc, nullptr, bgc2, nullptr, NTt,Dc,Dq, Dq,Dc,Dc,0, 1,1, 0,0,0,0,0,0,0,0, 1.f,0);
  k_gamma<<<4096, 256, 0, stream>>>(glog1, glogc, gammaf, NDl);
  // 5. serial gated recurrence (single-WGP WMMA scan, Wgh^T L2-resident)
  k_scan<<<1, 512, 0, stream>>>(zgf, zf, gammaf, bgh, wghT, hseqf);
  // 6. attention head projections
  gemmL(0, qbf, awqb, nullptr, qhb, abq, nullptr, NTt,Dc,Dc, Dc,Dc,Dc,0, 1,1, 0,0,0,0,0,0,0,0, 1.f,0);
  gemmL(0, mkb, awkb, nullptr, khb, abk, nullptr, NTM,Dc,Dc, Dc,Dc,Dc,0, 1,1, 0,0,0,0,0,0,0,0, 1.f,0);
  gemmL(0, mvb, awvb, nullptr, vhb, abv, nullptr, NTM,Dc,Dc, Dc,Dc,Dc,0, 1,1, 0,0,0,0,0,0,0,0, 1.f,0);
  // 7. scores[b,h,s,m] = qh . kh^T / sqrt(128)   (batched NT GEMM, z=(b*8+h))
  gemmL(1, qhb, khb, scoresf, nullptr, nullptr, nullptr,
        Sc, Mm, 128, Dc, Dc, Mm, 0, Bc*Hc, Hc,
        (long)Sc*Dc, 128,  (long)Mm*Dc, 128,
        (long)Hc*Sc*Mm, (long)Sc*Mm, 0, 0,
        0.08838834764831845f, 0);
  // 8. softmax + probs (bf16)
  k_softmax<<<Bc*Hc*Sc, 256, 0, stream>>>(scoresf, probsb, Mm);
  // 9. ctx[b,s,h,d] = probs @ vh   (batched NN GEMM)
  gemmL(0, probsb, vhb, ctxf, nullptr, nullptr, nullptr,
        Sc, 128, Mm, Mm, Dc, Dc, 0, Bc*Hc, Hc,
        (long)Hc*Sc*Mm, (long)Sc*Mm,  (long)Mm*Dc, 128,
        (long)Sc*Dc, 128, 0, 0, 1.f, 0);
  cvt(ctxf, ctxb, NDl);
  // 10. mem_out = ctx @ Awo + abo
  gemmL(0, ctxb, awob, mof, nullptr, abo, nullptr, NTt,Dc,Dc, Dc,Dc,Dc,0, 1,1, 0,0,0,0,0,0,0,0, 1.f,0);
  // 11. x_res = x + h_seq + mem_out + 0.1*v
  k_combine<<<4096, 256, 0, stream>>>(x, hseqf, mof, vf, xresf, NDl);
  // 12. FFN with fused residual -> d_out
  k_rmsnorm<<<NTt, 256, 0, stream>>>(xresf, n2s, xn2b, Dc);
  gemmL(0, xn2b, wf1b, nullptr, f1b, bf1, nullptr, NTt,Fc,Dc, Dc,Fc,Fc,0, 1,1, 0,0,0,0,0,0,0,0, 1.f,1);
  gemmL(0, f1b, wf2b, (float*)d_out, nullptr, bf2, xresf,
        NTt, Dc, Fc, Fc, Dc, Dc, Dc, 1,1, 0,0,0,0,0,0,0,0, 1.f, 0);
}
